// KmeansAttention_20452634264208
// MI455X (gfx1250) — compile-verified, matrix operand order checked by reading, not run on hardware
//
#include <hip/hip_runtime.h>
#include <hip/hip_bf16.h>

// ---------------- problem constants (match reference) ----------------
constexpr int B   = 4;
constexpr int H   = 8;
constexpr int T   = 4096;
constexpr int D   = 64;
constexpr int NC  = 64;
constexpr int WSZ = 128;
constexpr int MEM = 1;
constexpr int T2  = 2 * T;          // rows of concat(q,k)
constexpr int L   = MEM + WSZ;      // 129 kv rows per cluster
constexpr int LP  = 144;            // L padded to 16 (N-tiles of S)
constexpr int JP  = 160;            // L padded to 32 (K-tiles of P@V)
constexpr int XP  = 68;             // f32 LDS row pad (bank-conflict avoidance)

typedef __attribute__((ext_vector_type(16))) _Float16 v16h;
typedef __attribute__((ext_vector_type(8)))  _Float16 v8h;
typedef __attribute__((ext_vector_type(8)))  float    v8f;

// -------------------------------------------------------------------
// WMMA helpers (wave32, v_wmma_f32_16x16x32_f16).
// A-fragment (16x32 f16): lane = M%16 (two 16-lane groups share M),
//   halves j=0..15 map to K = j + (j/8)*8 + (lane/16)*8  -> two
//   contiguous 8-half (16B) runs at k0+(lane>>4)*8 and +16.
// B-fragment mirrors with N = lane%16, so loading row-major [N][K]
// data with the SAME loader contracts against the stored matrix
// transposed — what we need for Q·K^T, xn·means^T, and P·(V^T)^T.
// -------------------------------------------------------------------
__device__ __forceinline__ v16h load_frag16(const _Float16* base, int ld,
                                            int row0, int k0) {
  const int lane = threadIdx.x & 31;
  const _Float16* p = base + (size_t)(row0 + (lane & 15)) * ld
                           + k0 + ((lane >> 4) << 3);
  v8h lo = *(const v8h*)p;
  v8h hi = *(const v8h*)(p + 16);
  v16h r;
#pragma unroll
  for (int i = 0; i < 8; ++i) { r[i] = lo[i]; r[i + 8] = hi[i]; }
  return r;
}

__device__ __forceinline__ v8f wmma16(v16h a, v16h b, v8f c) {
  return __builtin_amdgcn_wmma_f32_16x16x32_f16(
      /*neg_a=*/false, a, /*neg_b=*/false, b,
      /*c_mod=*/(short)0, c, /*reuse_a=*/false, /*reuse_b=*/false);
}

// ---------------- kernel 0: zero accumulators ----------------
__global__ void kz_zero(float* __restrict__ out, float* __restrict__ denom,
                        float* __restrict__ aux) {
  const size_t N = (size_t)B * H * T * D;
  size_t i = (size_t)blockIdx.x * 256 + threadIdx.x;
  if (i < N) out[i] = 0.0f;
  if (i < (size_t)B * H * T) denom[i] = 0.0f;
  if (i == 0) aux[0] = 0.0f;
}

// ---------------- kernel 1: normalize + dists (WMMA) + argmax + aux --
// one block = 128 rows of x = concat(q,k) for one (b,h).
// dists[m][c] = xn[m]·means[c] computed on the matrix pipe; stored
// transposed [bh][c][t2] with coalesced 512B runs for the top-k reads.
__global__ void k_dists(const float* __restrict__ q, const float* __restrict__ k,
                        const float* __restrict__ means,
                        float* __restrict__ dists, float* __restrict__ aux) {
  extern __shared__ __align__(16) char smem[];
  float*    Mf = (float*)smem;            // [64][64]  f32 means (aux loss)
  float*    Xf = Mf + NC * D;             // [128][68] f32 xn (padded)
  float*    Sd = Xf + 128 * XP;           // [128][68] f32 dists staging
  _Float16* Mh = (_Float16*)(Sd + 128 * XP); // [64][64]  f16 means
  _Float16* Xh = Mh + NC * D;             // [128][64] f16 xn

  constexpr int CHUNKS = T2 / 128;        // 64
  const int bh = blockIdx.x / CHUNKS;
  const int t0 = (blockIdx.x % CHUNKS) * 128;
  const int h  = bh & (H - 1);
  const int tid = threadIdx.x;

  for (int i = tid; i < NC * D; i += 256) {
    float m = means[(size_t)h * NC * D + i];
    Mf[i] = m; Mh[i] = (_Float16)m;
  }
  const float* src_base = (t0 < T) ? (q + ((size_t)bh * T + t0) * D)
                                   : (k + ((size_t)bh * T + (t0 - T)) * D);
  for (int e = tid; e < 128 * (D / 4); e += 256) {
    const int row = e >> 4, c4 = (e & 15) << 2;
    *(float4*)(Xf + row * XP + c4) = *(const float4*)(src_base + (size_t)row * D + c4);
  }
  __syncthreads();

  if (tid < 128) {                        // per-row L2 normalize
    float nrm = 0.0f;
#pragma unroll
    for (int d = 0; d < D; ++d) { float x = Xf[tid * XP + d]; nrm += x * x; }
    const float inv = 1.0f / fmaxf(sqrtf(nrm), 1e-12f);
#pragma unroll
    for (int d = 0; d < D; ++d) {
      float x = Xf[tid * XP + d] * inv;
      Xf[tid * XP + d] = x;
      Xh[tid * D + d] = (_Float16)x;
    }
  }
  __syncthreads();

  // ---- dists = xn · means^T on the WMMA pipe (per wave: 16 rows) ----
  const int w = tid >> 5, lane = tid & 31, colh = lane & 15;
  v16h a0 = load_frag16(Xh, D, w * 16, 0);
  v16h a1 = load_frag16(Xh, D, w * 16, 32);
#pragma unroll
  for (int ct = 0; ct < 4; ++ct) {
    v16h b0 = load_frag16(Mh, D, ct * 16, 0);
    v16h b1 = load_frag16(Mh, D, ct * 16, 32);
    v8f acc = {0.f, 0.f, 0.f, 0.f, 0.f, 0.f, 0.f, 0.f};
    acc = wmma16(a0, b0, acc);
    acc = wmma16(a1, b1, acc);
#pragma unroll
    for (int r = 0; r < 8; ++r) {
      const int row = w * 16 + r + ((lane >> 4) << 3);
      Sd[row * XP + ct * 16 + colh] = acc[r];
    }
  }
  __syncthreads();

  if (tid < 128) {                        // argmax + aux loss (f32)
    int best = 0; float bv = -1e30f;
    for (int c = 0; c < NC; ++c) {
      float s = Sd[tid * XP + c];
      if (s > bv) { bv = s; best = c; }
    }
    float e = 0.0f;
#pragma unroll
    for (int d = 0; d < D; ++d) {
      float dd = Xf[tid * XP + d] - Mf[best * D + d];
      e += dd * dd;
    }
    atomicAdd(aux, e);
  }
  for (int e = tid; e < NC * 128; e += 256) {   // transposed coalesced store
    const int c = e >> 7, i = e & 127;
    dists[((size_t)bh * NC + c) * T2 + t0 + i] = Sd[i * XP + c];
  }
}

// ---------------- kernel 2: exact per-cluster top-128 ----------------
// bitonic sort (descending) of 4096 packed keys; ties -> smaller index
// first (matches jax.lax.top_k). grid: (B*H*NC, 2)  y==1 -> kv side.
__global__ void k_topk(const float* __restrict__ dists,
                       int* __restrict__ qidx, int* __restrict__ kvidx) {
  __shared__ unsigned long long keys[T];   // 32 KB
  const int  bhc = blockIdx.x;
  const bool isK = (blockIdx.y != 0);
  const float* col = dists + (size_t)bhc * T2 + (isK ? T : 0);

  for (int i = threadIdx.x; i < T; i += 256) {
    unsigned u = __float_as_uint(col[i]);
    u = (u & 0x80000000u) ? ~u : (u | 0x80000000u);    // order-preserving map
    keys[i] = ((unsigned long long)u << 32) | (unsigned)(0xFFFFFFFFu - i);
  }
  __syncthreads();

  for (int kk = 2; kk <= T; kk <<= 1) {
    for (int j = kk >> 1; j > 0; j >>= 1) {
      for (int i = threadIdx.x; i < T; i += 256) {
        const int ixj = i ^ j;
        if (ixj > i) {
          const bool desc = ((i & kk) == 0);
          unsigned long long a = keys[i], b = keys[ixj];
          if ((a < b) == desc) { keys[i] = b; keys[ixj] = a; }
        }
      }
      __syncthreads();
    }
  }
  int* dst = (isK ? kvidx : qidx) + (size_t)bhc * WSZ;
  if (threadIdx.x < WSZ)
    dst[threadIdx.x] = (int)(0xFFFFFFFFu - (unsigned)keys[threadIdx.x]);
}

// ---------------- kernel 3: gathered attention (WMMA) ----------------
// one block (8 waves, 256 thr) per (b,h,cluster): S = Qg·Kg^T (128x129),
// softmax, O = P·Vg (128x64), atomic scatter of numer + denom count.
__global__ void k_attn(const float* __restrict__ q, const float* __restrict__ k,
                       const float* __restrict__ v,
                       const float* __restrict__ mem_key,
                       const float* __restrict__ mem_value,
                       const int* __restrict__ qidx, const int* __restrict__ kvidx,
                       float* __restrict__ out, float* __restrict__ denom) {
  extern __shared__ __align__(16) char smem[];
  _Float16* Qh = (_Float16*)smem;          // [128][64]
  _Float16* Kh = Qh + WSZ * D;             // [144][64] rows>=129 zero
  _Float16* VT = Kh + LP * D;              // [64][160]  V^T, cols>=129 zero
  _Float16* Ph = VT + D * JP;              // [128][160] cols>=144 zero
  int* qi_s = (int*)(Ph + WSZ * JP);
  int* ki_s = qi_s + WSZ;

  const int bhc = blockIdx.x;
  const int bh  = bhc / NC;
  const int c   = bhc % NC;
  const int h   = bh & (H - 1);
  const int tid = threadIdx.x;
  const float scale = 0.125f;              // D^-0.5

  if (tid < WSZ) {
    qi_s[tid] = qidx[(size_t)bhc * WSZ + tid];
    ki_s[tid] = kvidx[(size_t)bhc * WSZ + tid];
  }
  __syncthreads();

  // ---- gather q -> Qh (f16), {mem_key|k} -> Kh, {mem_value|v} -> VT ----
  for (int e = tid; e < WSZ * (D / 4); e += 256) {
    const int row = e >> 4, c4 = (e & 15) << 2;
    const float4 f = *(const float4*)(q + ((size_t)bh * T + qi_s[row]) * D + c4);
    _Float16* dst = Qh + row * D + c4;
    dst[0] = (_Float16)f.x; dst[1] = (_Float16)f.y;
    dst[2] = (_Float16)f.z; dst[3] = (_Float16)f.w;
  }
  for (int e = tid; e < L * (D / 4); e += 256) {
    const int row = e >> 4, c4 = (e & 15) << 2;
    const float* kp; const float* vp;
    if (row == 0) {
      kp = mem_key   + (size_t)(h * NC + c) * MEM * D;
      vp = mem_value + (size_t)(h * NC + c) * MEM * D;
    } else {
      const size_t base = ((size_t)bh * T + ki_s[row - 1]) * D;
      kp = k + base; vp = v + base;
    }
    const float4 fk = *(const float4*)(kp + c4);
    const float4 fv = *(const float4*)(vp + c4);
    _Float16* kd = Kh + row * D + c4;
    kd[0] = (_Float16)fk.x; kd[1] = (_Float16)fk.y;
    kd[2] = (_Float16)fk.z; kd[3] = (_Float16)fk.w;
    VT[(c4 + 0) * JP + row] = (_Float16)fv.x;
    VT[(c4 + 1) * JP + row] = (_Float16)fv.y;
    VT[(c4 + 2) * JP + row] = (_Float16)fv.z;
    VT[(c4 + 3) * JP + row] = (_Float16)fv.w;
  }
  // zero pads: Kh rows 129..143, VT cols 129..159, Ph cols 144..159
  for (int i = tid; i < (LP - L) * D; i += 256)
    Kh[(L + (i >> 6)) * D + (i & 63)] = (_Float16)0.0f;
  for (int i = tid; i < D * (JP - L); i += 256)
    VT[(i / (JP - L)) * JP + L + (i % (JP - L))] = (_Float16)0.0f;
  for (int i = tid; i < WSZ * (JP - LP); i += 256)
    Ph[(i >> 4) * JP + LP + (i & 15)] = (_Float16)0.0f;
  __syncthreads();

  // ---- S = Qg · Kg^T  (per wave: 16 rows x 144 cols, 9 N-tiles) ----
  const int w    = tid >> 5;
  const int lane = tid & 31;
  const int colh = lane & 15;
  v16h aq0 = load_frag16(Qh, D, w * 16, 0);
  v16h aq1 = load_frag16(Qh, D, w * 16, 32);
  v8f S[9];
#pragma unroll
  for (int ct = 0; ct < 9; ++ct) {
    v16h b0 = load_frag16(Kh, D, ct * 16, 0);
    v16h b1 = load_frag16(Kh, D, ct * 16, 32);
    v8f acc = {0.f, 0.f, 0.f, 0.f, 0.f, 0.f, 0.f, 0.f};
    acc = wmma16(aq0, b0, acc);
    acc = wmma16(aq1, b1, acc);
    S[ct] = acc;
  }

  // ---- softmax over 129 valid cols; store unnormalized P (f16) ----
  float rs[8];
#pragma unroll
  for (int r = 0; r < 8; ++r) {
    float m = -1e30f;
#pragma unroll
    for (int ct = 0; ct < 9; ++ct) {
      float s = S[ct][r] * scale;
      if (ct == 8 && colh != 0) s = -1e30f;   // cols 129..143 invalid
      S[ct][r] = s;
      m = fmaxf(m, s);
    }
#pragma unroll
    for (int off = 1; off < 16; off <<= 1) m = fmaxf(m, __shfl_xor(m, off, 16));
    float sum = 0.0f;
#pragma unroll
    for (int ct = 0; ct < 9; ++ct) {
      float p = __expf(S[ct][r] - m);
      S[ct][r] = p;
      sum += p;
    }
#pragma unroll
    for (int off = 1; off < 16; off <<= 1) sum += __shfl_xor(sum, off, 16);
    rs[r] = sum;
    const int row = w * 16 + r + ((lane >> 4) << 3);
#pragma unroll
    for (int ct = 0; ct < 9; ++ct)
      Ph[row * JP + ct * 16 + colh] = (_Float16)S[ct][r];
  }
  __syncthreads();

  // ---- O = P · V  (contract j over 5 K-tiles of 32) + scatter ----
#pragma unroll
  for (int ct2 = 0; ct2 < 4; ++ct2) {
    v8f o = {0.f, 0.f, 0.f, 0.f, 0.f, 0.f, 0.f, 0.f};
#pragma unroll
    for (int jt = 0; jt < 5; ++jt) {
      v16h ap = load_frag16(Ph, JP, w * 16, jt * 32);
      v16h bv = load_frag16(VT, JP, ct2 * 16, jt * 32);
      o = wmma16(ap, bv, o);
    }
#pragma unroll
    for (int r = 0; r < 8; ++r) {
      const int row = w * 16 + r + ((lane >> 4) << 3);
      const int t   = qi_s[row];
      const float val = o[r] / rs[r];
      atomicAdd(&out[(((size_t)bh * T + t) * D) + ct2 * 16 + colh], val);
    }
  }
  if (tid < WSZ) atomicAdd(&denom[(size_t)bh * T + qi_s[tid]], 1.0f);
}

// ---------------- kernel 4: normalize + aux loss ----------------
__global__ void k_final(float* __restrict__ out, const float* __restrict__ denom,
                        const float* __restrict__ aux) {
  const size_t N = (size_t)B * H * T * D;
  size_t i = (size_t)blockIdx.x * 256 + threadIdx.x;
  if (i < N) out[i] = out[i] / (denom[i >> 6] + 1e-5f);
  if (i == 0) out[N] = aux[0] * (1.0f / ((float)B * H * T2 * D));
}

// -------------------------------------------------------------------
extern "C" void kernel_launch(void* const* d_in, const int* in_sizes, int n_in,
                              void* d_out, int out_size, void* d_ws, size_t ws_size,
                              hipStream_t stream) {
  const float* q         = (const float*)d_in[0];
  const float* k         = (const float*)d_in[1];
  const float* v         = (const float*)d_in[2];
  const float* means     = (const float*)d_in[3];
  const float* mem_key   = (const float*)d_in[4];
  const float* mem_value = (const float*)d_in[5];
  float* out = (float*)d_out;

  // workspace carve (all 16B aligned sizes)
  char* ws = (char*)d_ws;
  float* dists = (float*)ws;                                   // BH*NC*T2 f32 = 64 MB
  size_t off = (size_t)B * H * NC * T2 * sizeof(float);
  int* qidx  = (int*)(ws + off); off += (size_t)B * H * NC * WSZ * sizeof(int);
  int* kvidx = (int*)(ws + off); off += (size_t)B * H * NC * WSZ * sizeof(int);
  float* denom = (float*)(ws + off); off += (size_t)B * H * T * sizeof(float);
  float* aux   = (float*)(ws + off);

  const size_t N = (size_t)B * H * T * D;
  const int zb = (int)((N + 255) / 256);
  kz_zero<<<zb, 256, 0, stream>>>(out, denom, aux);

  const size_t dists_lds = (size_t)(NC * D + 128 * XP + 128 * XP) * sizeof(float)
                         + (size_t)(NC * D + 128 * D) * sizeof(_Float16); // 110,592 B
  k_dists<<<B * H * (T2 / 128), 256, dists_lds, stream>>>(q, k, means, dists, aux);

  k_topk<<<dim3(B * H * NC, 2), 256, 0, stream>>>(dists, qidx, kvidx);

  const size_t attn_lds = (size_t)(WSZ * D + LP * D + D * JP + WSZ * JP) * sizeof(_Float16)
                        + 2 * WSZ * sizeof(int);   // 97,280 B < 320 KB/WGP
  k_attn<<<B * H * NC, 256, attn_lds, stream>>>(q, k, v, mem_key, mem_value,
                                                qidx, kvidx, out, denom);

  k_final<<<zb, 256, 0, stream>>>(out, denom, aux);
}
